// GMT_12317966204983
// MI455X (gfx1250) — matrix operand order repaired and checked by reading, not compiled, not run on hardware
//
#include <hip/hip_runtime.h>
#include <cmath>

typedef __attribute__((ext_vector_type(16))) _Float16 v16h;
typedef __attribute__((ext_vector_type(8)))  float    v8f;

#define HC 32   // heads * channels
#define NH 2    // heads
#define CH 16   // channels per head

// ---------------------------------------------------------------------------
// utility fills
// ---------------------------------------------------------------------------
__global__ void fill_f32(float* __restrict__ p, long long n, float v) {
  long long i = (long long)blockIdx.x * blockDim.x + threadIdx.x;
  if (i < n) p[i] = v;
}

// pack {Wq|Wk|Wv|Ws} (each [K,32] f32 row-major) into Wc [K,128] f16 row-major,
// and biases into bc[128].
__global__ void pack_weights(const float* __restrict__ Wq, const float* __restrict__ bq,
                             const float* __restrict__ Wk, const float* __restrict__ bk,
                             const float* __restrict__ Wv, const float* __restrict__ bv,
                             const float* __restrict__ Ws, const float* __restrict__ bs,
                             int K, _Float16* __restrict__ Wc, float* __restrict__ bc) {
  int idx = blockIdx.x * blockDim.x + threadIdx.x;
  if (idx < K * 128) {
    int k = idx >> 7, j = idx & 127;
    int grp = j >> 5, jj = j & 31;
    const float* W = (grp == 0) ? Wq : (grp == 1) ? Wk : (grp == 2) ? Wv : Ws;
    Wc[idx] = (_Float16)W[k * 32 + jj];
  }
  if (idx < 128) {
    int grp = idx >> 5, jj = idx & 31;
    const float* b = (grp == 0) ? bq : (grp == 1) ? bk : (grp == 2) ? bv : bs;
    bc[idx] = b[jj];
  }
}

// ---------------------------------------------------------------------------
// Fused projection GEMM:  Y[N,128] = X[N,K] @ Wc[K,128] + bc   (K = 128 or 32)
// Wave tile: 16 rows x 32 cols = two v_wmma_f32_16x16x32_f16 accumulators.
// Block = 256 threads = 8 waves = 128 rows; grid.y covers 4 col-pairs.
// ---------------------------------------------------------------------------
__global__ void __launch_bounds__(256)
gemm_qkvs_wmma(const float* __restrict__ X, int nrows, int K,
               const _Float16* __restrict__ W, const float* __restrict__ bias,
               float* __restrict__ Y) {
  const int lane = threadIdx.x & 31;
  const int wave = threadIdx.x >> 5;
  const int hi   = lane >> 4;                 // 0: lanes 0-15, 1: lanes 16-31
  const int rowBase = blockIdx.x * 128 + wave * 16;
  const int colBase = blockIdx.y * 32;
  const int n0 = colBase + (lane & 15);

  int arow = rowBase + (lane & 15);
  if (arow >= nrows) arow = nrows - 1;        // clamp loads; stores are masked
  const float* xrow = X + (size_t)arow * (size_t)K;

  v8f c0 = {}; v8f c1 = {};

  for (int k0 = 0; k0 < K; k0 += 32) {
    // ---- A fragment: X[arow][k0 + hi*8 + {0..7}] and [... + 16 + {0..7}]
    const int bk = k0 + hi * 8;
    float4 f0 = *(const float4*)(xrow + bk);
    float4 f1 = *(const float4*)(xrow + bk + 4);
    float4 f2 = *(const float4*)(xrow + bk + 16);
    float4 f3 = *(const float4*)(xrow + bk + 20);
    v16h a;
    a[0] = (_Float16)f0.x;  a[1] = (_Float16)f0.y;  a[2]  = (_Float16)f0.z;  a[3]  = (_Float16)f0.w;
    a[4] = (_Float16)f1.x;  a[5] = (_Float16)f1.y;  a[6]  = (_Float16)f1.z;  a[7]  = (_Float16)f1.w;
    a[8] = (_Float16)f2.x;  a[9] = (_Float16)f2.y;  a[10] = (_Float16)f2.z;  a[11] = (_Float16)f2.w;
    a[12] = (_Float16)f3.x; a[13] = (_Float16)f3.y; a[14] = (_Float16)f3.z;  a[15] = (_Float16)f3.w;

    // ---- B fragments: W[k0 + hi*8 + (j&7) + (j>>3)*16][n]  (column-strided; W is L2-resident)
    v16h b0, b1;
    #pragma unroll
    for (int j = 0; j < 16; ++j) {
      const int kk = k0 + hi * 8 + (j & 7) + ((j >> 3) << 4);
      b0[j] = W[(size_t)kk * 128 + n0];
      b1[j] = W[(size_t)kk * 128 + n0 + 16];
    }

    c0 = __builtin_amdgcn_wmma_f32_16x16x32_f16(false, a, false, b0, (short)0, c0, false, false);
    c1 = __builtin_amdgcn_wmma_f32_16x16x32_f16(false, a, false, b1, (short)0, c1, false, false);
  }

  // ---- store: C/D layout VGPR i -> M = i + hi*8, N = lane%16
  const float bias0 = bias[n0];
  const float bias1 = bias[n0 + 16];
  #pragma unroll
  for (int i = 0; i < 8; ++i) {
    const int r = rowBase + i + hi * 8;
    if (r < nrows) {
      Y[(size_t)r * 128 + n0]      = c0[i] + bias0;
      Y[(size_t)r * 128 + n0 + 16] = c1[i] + bias1;
    }
  }
}

// ---------------------------------------------------------------------------
// Edge phase.  qkvs layout per node: [q(0:32) | k(32:64) | v(64:96) | skip(96:128)]
// ---------------------------------------------------------------------------
__device__ __forceinline__ void atomicMaxFloat(float* addr, float v) {
  // sign-aware single-atomic float max (init must be -inf = 0xFF800000)
  if (v >= 0.0f) atomicMax((int*)addr, __float_as_int(v));
  else           atomicMin((unsigned int*)addr, __float_as_uint(v));
}

__global__ void edge_score_max(const int* __restrict__ ei, long long E,
                               const float* __restrict__ qkvs,
                               float* __restrict__ score, float* __restrict__ m) {
  long long e = (long long)blockIdx.x * blockDim.x + threadIdx.x;
  if (e >= E) return;
  const int src = ei[e];
  const int dst = ei[E + e];
  const float* qd = qkvs + (size_t)dst * 128;        // q
  const float* ks = qkvs + (size_t)src * 128 + 32;   // k
  #pragma unroll
  for (int h = 0; h < NH; ++h) {
    const float4* q4 = (const float4*)(qd + h * CH);
    const float4* k4 = (const float4*)(ks + h * CH);
    float s = 0.0f;
    #pragma unroll
    for (int i = 0; i < 4; ++i) {
      float4 qa = q4[i], kb = k4[i];
      s += qa.x * kb.x + qa.y * kb.y + qa.z * kb.z + qa.w * kb.w;
    }
    s *= 0.25f;                                       // 1/sqrt(16)
    score[e * NH + h] = s;
    atomicMaxFloat(&m[(size_t)dst * NH + h], s);
  }
}

__global__ void edge_exp_sum(const int* __restrict__ ei, long long E,
                             float* __restrict__ score, const float* __restrict__ m,
                             float* __restrict__ denom) {
  long long e = (long long)blockIdx.x * blockDim.x + threadIdx.x;
  if (e >= E) return;
  const int dst = ei[E + e];
  #pragma unroll
  for (int h = 0; h < NH; ++h) {
    float ex = expf(score[e * NH + h] - m[(size_t)dst * NH + h]);
    score[e * NH + h] = ex;                           // reuse buffer for exp(score)
    atomicAdd(&denom[(size_t)dst * NH + h], ex);
  }
}

__global__ void edge_aggregate(const int* __restrict__ ei, long long E,
                               const float* __restrict__ qkvs,
                               const float* __restrict__ score,
                               const float* __restrict__ denom,
                               float* __restrict__ agg) {
  long long e = (long long)blockIdx.x * blockDim.x + threadIdx.x;
  if (e >= E) return;
  const int src = ei[e];
  const int dst = ei[E + e];
  #pragma unroll
  for (int h = 0; h < NH; ++h) {
    const float alpha = score[e * NH + h] / denom[(size_t)dst * NH + h];
    const float* v = qkvs + (size_t)src * 128 + 64 + h * CH;
    float* o = agg + (size_t)dst * HC + h * CH;
    #pragma unroll
    for (int c = 0; c < CH; ++c) atomicAdd(&o[c], alpha * v[c]);
  }
}

__global__ void combine_relu(const float* __restrict__ qkvs, const float* __restrict__ agg,
                             float* __restrict__ hout, long long n32) {
  long long i = (long long)blockIdx.x * blockDim.x + threadIdx.x;
  if (i >= n32) return;
  const long long n = i >> 5;
  const int c = (int)(i & 31);
  float t = agg[i] + qkvs[n * 128 + 96 + c];          // aggregated msgs + skip
  hout[i] = t > 0.0f ? t : 0.0f;
}

// ---------------------------------------------------------------------------
// Readout: mean pool per graph -> MLP -> log_softmax
// ---------------------------------------------------------------------------
__global__ void pool_accum(const float* __restrict__ h, const int* __restrict__ batch,
                           long long n32, float* __restrict__ pool, float* __restrict__ cnt) {
  long long i = (long long)blockIdx.x * blockDim.x + threadIdx.x;
  if (i >= n32) return;
  const long long n = i >> 5;
  const int c = (int)(i & 31);
  const int b = batch[n];
  atomicAdd(&pool[(size_t)b * HC + c], h[i]);
  if (c == 0) atomicAdd(&cnt[b], 1.0f);
}

__global__ void head_mlp(const float* __restrict__ pool, const float* __restrict__ cnt,
                         const float* __restrict__ Wf1, const float* __restrict__ bf1,
                         const float* __restrict__ Wf2, const float* __restrict__ bf2,
                         float* __restrict__ out, int G, int ncls) {
  int g = blockIdx.x * blockDim.x + threadIdx.x;
  if (g >= G) return;
  float c = cnt[g]; if (c < 1.0f) c = 1.0f;
  float gv[HC];
  #pragma unroll
  for (int i = 0; i < HC; ++i) gv[i] = pool[(size_t)g * HC + i] / c;
  float hid[CH];
  #pragma unroll
  for (int j = 0; j < CH; ++j) {
    float s = bf1[j];
    #pragma unroll
    for (int i = 0; i < HC; ++i) s += gv[i] * Wf1[i * CH + j];
    hid[j] = s > 0.0f ? s : 0.0f;
  }
  float lg[16];
  float mx = -3.0e38f;
  for (int k = 0; k < ncls; ++k) {
    float s = bf2[k];
    #pragma unroll
    for (int j = 0; j < CH; ++j) s += hid[j] * Wf2[j * ncls + k];
    lg[k] = s;
    if (s > mx) mx = s;
  }
  float sum = 0.0f;
  for (int k = 0; k < ncls; ++k) sum += expf(lg[k] - mx);
  const float lse = mx + logf(sum);
  for (int k = 0; k < ncls; ++k) out[(size_t)g * ncls + k] = lg[k] - lse;
}

// ---------------------------------------------------------------------------
// host driver
// ---------------------------------------------------------------------------
static void launch_layer(const float* X, int K, const _Float16* Wc, const float* bc,
                         const int* ei, long long N, long long E,
                         float* qkvs, float* score, float* m, float* denom, float* agg,
                         float* hout, hipStream_t stream) {
  dim3 ggrid((unsigned)((N + 127) / 128), 4);
  gemm_qkvs_wmma<<<ggrid, 256, 0, stream>>>(X, (int)N, K, Wc, bc, qkvs);

  fill_f32<<<(unsigned)((N * NH + 255) / 256), 256, 0, stream>>>(m, N * NH, -INFINITY);
  hipMemsetAsync(denom, 0, (size_t)N * NH * sizeof(float), stream);

  const unsigned eb = (unsigned)((E + 255) / 256);
  edge_score_max<<<eb, 256, 0, stream>>>(ei, E, qkvs, score, m);
  edge_exp_sum  <<<eb, 256, 0, stream>>>(ei, E, score, m, denom);

  hipMemsetAsync(agg, 0, (size_t)N * HC * sizeof(float), stream);
  edge_aggregate<<<eb, 256, 0, stream>>>(ei, E, qkvs, score, denom, agg);

  combine_relu<<<(unsigned)((N * HC + 255) / 256), 256, 0, stream>>>(qkvs, agg, hout, N * HC);
}

extern "C" void kernel_launch(void* const* d_in, const int* in_sizes, int n_in,
                              void* d_out, int out_size, void* d_ws, size_t ws_size,
                              hipStream_t stream) {
  const float* x     = (const float*)d_in[0];
  const int*   ei    = (const int*)d_in[1];
  const int*   batch = (const int*)d_in[2];
  const float* Wq1 = (const float*)d_in[3],  *bq1 = (const float*)d_in[4];
  const float* Wk1 = (const float*)d_in[5],  *bk1 = (const float*)d_in[6];
  const float* Wv1 = (const float*)d_in[7],  *bv1 = (const float*)d_in[8];
  const float* Ws1 = (const float*)d_in[9],  *bs1 = (const float*)d_in[10];
  const float* Wq2 = (const float*)d_in[11], *bq2 = (const float*)d_in[12];
  const float* Wk2 = (const float*)d_in[13], *bk2 = (const float*)d_in[14];
  const float* Wv2 = (const float*)d_in[15], *bv2 = (const float*)d_in[16];
  const float* Ws2 = (const float*)d_in[17], *bs2 = (const float*)d_in[18];
  const float* Wf1 = (const float*)d_in[19], *bf1 = (const float*)d_in[20];
  const float* Wf2 = (const float*)d_in[21], *bf2 = (const float*)d_in[22];

  const long long N    = in_sizes[2];
  const long long E    = in_sizes[1] / 2;
  const int       FIN  = (int)(in_sizes[0] / N);      // 128
  const int       NCLS = in_sizes[22];                // 10
  const int       G    = out_size / NCLS;             // 256

  // workspace carve-up (f32 units)
  float* ws = (float*)d_ws;
  size_t off = 0;
  float* qkvs  = ws + off; off += (size_t)N * 128;
  float* h1    = ws + off; off += (size_t)N * HC;
  float* h2    = ws + off; off += (size_t)N * HC;
  float* score = ws + off; off += (size_t)E * NH;
  float* m     = ws + off; off += (size_t)N * NH;
  float* denom = ws + off; off += (size_t)N * NH;
  float* agg   = ws + off; off += (size_t)N * HC;
  float* poolb = ws + off; off += (size_t)G * HC;
  float* cntb  = ws + off; off += (size_t)G;
  off = (off + 3) & ~(size_t)3;
  _Float16* Wc1 = (_Float16*)(ws + off); off += (size_t)(128 * 128) / 2;
  _Float16* Wc2 = (_Float16*)(ws + off); off += (size_t)(HC * 128) / 2;
  float* bc1 = ws + off; off += 128;
  float* bc2 = ws + off; off += 128;
  (void)ws_size; (void)n_in;

  // pack weights (deterministic, every call)
  pack_weights<<<(FIN * 128 + 255) / 256, 256, 0, stream>>>(Wq1, bq1, Wk1, bk1, Wv1, bv1, Ws1, bs1, FIN, Wc1, bc1);
  pack_weights<<<(HC  * 128 + 255) / 256, 256, 0, stream>>>(Wq2, bq2, Wk2, bk2, Wv2, bv2, Ws2, bs2, HC,  Wc2, bc2);

  // layer 1: x[N,128] -> h1[N,32]
  launch_layer(x,  FIN, Wc1, bc1, ei, N, E, qkvs, score, m, denom, agg, h1, stream);
  // layer 2: h1[N,32] -> h2[N,32]
  launch_layer(h1, HC,  Wc2, bc2, ei, N, E, qkvs, score, m, denom, agg, h2, stream);

  // global mean pool + head
  hipMemsetAsync(poolb, 0, (size_t)G * HC * sizeof(float), stream);
  hipMemsetAsync(cntb,  0, (size_t)G * sizeof(float), stream);
  pool_accum<<<(unsigned)((N * HC + 255) / 256), 256, 0, stream>>>(h2, batch, N * HC, poolb, cntb);
  head_mlp<<<(G + 255) / 256, 256, 0, stream>>>(poolb, cntb, Wf1, bf1, Wf2, bf2, (float*)d_out, G, NCLS);
}
